// LlamaAttentionExperimental_84937273246104
// MI455X (gfx1250) — matrix operand order repaired and checked by reading, not compiled, not run on hardware
//
#include <hip/hip_runtime.h>
#include <cstdint>
#include <cstddef>

#define HIDDEN 2048
#define NH     32
#define NKV    8
#define HD     64
#define SEQ    2048
#define BATCH  2
#define KEEP   ((SEQ * 4) / 10)   /* int(2048*0.4) = 819 */
#define SCALE  0.125f             /* 1/sqrt(64) */
#define NEGF   (-1.0e30f)

typedef __attribute__((ext_vector_type(16))) __bf16 v16bf;
typedef __attribute__((ext_vector_type(8)))  float  v8f;

union FragU { uint4 q[2]; v16bf v; };

// ---------------------------------------------------------------- helpers
static __device__ __forceinline__ unsigned short f2bf(float f) {
  union { float f; unsigned u; } a; a.f = f;
  unsigned u = a.u + 0x7FFFu + ((a.u >> 16) & 1u);   // round-to-nearest-even
  return (unsigned short)(u >> 16);
}

static __device__ __forceinline__ v8f wmma_bf16(v16bf a, v16bf b, v8f c) {
  return __builtin_amdgcn_wmma_f32_16x16x32_bf16(false, a, false, b, (short)0, c,
                                                 false, false);
}

// A operand (global): rows row0..+15, K-slice col0..+31 of row-major [*, ld]
// lane L: row = row0 + (L&15); chunks at +0B and +32B (both 16B-aligned)
static __device__ __forceinline__ v16bf frag_a(const unsigned short* base, int ld,
                                               int row0, int col0) {
  int lane = threadIdx.x & 31;
  const unsigned short* p =
      base + (size_t)(row0 + (lane & 15)) * ld + col0 + ((lane >> 4) << 3);
  FragU f;
  f.q[0] = *(const uint4*)(p);
  f.q[1] = *(const uint4*)(p + 16);
  return f.v;
}

// B operand (global) for X*Y^T: Y rows n0..+15 become B columns; K-slice k0..+31
// lane L: chunks at +0B and +32B (both 16B-aligned), same k order as frag_a
static __device__ __forceinline__ v16bf frag_b(const unsigned short* base, int ld,
                                               int n0, int k0) {
  int lane = threadIdx.x & 31;
  const unsigned short* p =
      base + (size_t)(n0 + (lane & 15)) * ld + k0 + ((lane >> 4) << 3);
  FragU f;
  f.q[0] = *(const uint4*)(p);
  f.q[1] = *(const uint4*)(p + 16);
  return f.v;
}

// ---------------------------------------------------------------- kernels
__global__ void f32_to_bf16_k(const float* __restrict__ in,
                              unsigned short* __restrict__ out, int n) {
  int i = blockIdx.x * 256 + threadIdx.x;
  if (i < n) out[i] = f2bf(in[i]);
}

// C(MxN) = A(MxK,bf16) * B(NxK,bf16)^T.
// 4-wave workgroup -> 128x64 block tile; per K-step of 32 the block stages
// A(128x32) + B(64x32) into LDS with global_load_async_to_lds_b128 (ASYNCcnt),
// then each wave computes a 32x64 sub-tile (8 WMMAs) from ds_load_b128 fragments.
__global__ __launch_bounds__(128) void gemm_xyt_lds(
    const unsigned short* __restrict__ A, const unsigned short* __restrict__ Bm,
    float* __restrict__ C, int M, int N, int K) {
  __shared__ unsigned short ldsA[128 * 32];
  __shared__ unsigned short ldsB[64 * 32];
  int tid = threadIdx.x;
  int w = tid >> 5, lane = tid & 31, half = lane >> 4, nl = lane & 15;
  int n0 = blockIdx.x * 64, m0 = blockIdx.y * 128;

  unsigned ldsA0 = (unsigned)(size_t)(void*)&ldsA[0];
  unsigned ldsB0 = (unsigned)(size_t)(void*)&ldsB[0];
  unsigned long long aBase = (unsigned long long)(size_t)A;
  unsigned long long bBase = (unsigned long long)(size_t)Bm;

  v8f zero{};
  v8f acc[2][4];
#pragma unroll
  for (int mi = 0; mi < 2; ++mi)
#pragma unroll
    for (int nt = 0; nt < 4; ++nt) acc[mi][nt] = zero;

  for (int k0 = 0; k0 < K; k0 += 32) {
    // ---- stage A tile: 128 rows x 32 cols bf16 = 512 x 16B chunks
#pragma unroll
    for (int j = 0; j < 4; ++j) {
      int id = tid + j * 128;
      int row = id >> 2, c4 = id & 3;
      unsigned voff = (unsigned)(((size_t)(m0 + row) * K + k0) * 2) + c4 * 16;
      unsigned laddr = ldsA0 + row * 64 + c4 * 16;
      asm volatile("global_load_async_to_lds_b128 %0, %1, %2"
                   :: "v"(laddr), "v"(voff), "s"(aBase) : "memory");
    }
    // ---- stage B tile: 64 rows x 32 cols bf16 = 256 x 16B chunks
#pragma unroll
    for (int j = 0; j < 2; ++j) {
      int id = tid + j * 128;
      int row = id >> 2, c4 = id & 3;
      unsigned voff = (unsigned)(((size_t)(n0 + row) * K + k0) * 2) + c4 * 16;
      unsigned laddr = ldsB0 + row * 64 + c4 * 16;
      asm volatile("global_load_async_to_lds_b128 %0, %1, %2"
                   :: "v"(laddr), "v"(voff), "s"(bBase) : "memory");
    }
    asm volatile("s_wait_asynccnt 0x0" ::: "memory");
    __syncthreads();

    // ---- fragments from LDS (ds_load_b128 path)
    FragU fa0, fa1;
    int r0 = w * 32 + (lane & 15);
    fa0.q[0] = *(const uint4*)&ldsA[r0 * 32 + half * 8];
    fa0.q[1] = *(const uint4*)&ldsA[r0 * 32 + half * 8 + 16];
    fa1.q[0] = *(const uint4*)&ldsA[(r0 + 16) * 32 + half * 8];
    fa1.q[1] = *(const uint4*)&ldsA[(r0 + 16) * 32 + half * 8 + 16];
#pragma unroll
    for (int nt = 0; nt < 4; ++nt) {
      FragU fb;
      fb.q[0] = *(const uint4*)&ldsB[(nt * 16 + nl) * 32 + half * 8];
      fb.q[1] = *(const uint4*)&ldsB[(nt * 16 + nl) * 32 + half * 8 + 16];
      acc[0][nt] = wmma_bf16(fa0.v, fb.v, acc[0][nt]);
      acc[1][nt] = wmma_bf16(fa1.v, fb.v, acc[1][nt]);
    }
    __syncthreads();
  }
#pragma unroll
  for (int mi = 0; mi < 2; ++mi)
#pragma unroll
    for (int nt = 0; nt < 4; ++nt)
#pragma unroll
      for (int r = 0; r < 8; ++r) {
        int mr = m0 + w * 32 + mi * 16 + r + half * 8;
        C[(size_t)mr * N + n0 + nt * 16 + nl] = acc[mi][nt][r];
      }
}

// RoPE on Q (fp32, [b,s,h*64+d]) -> bf16 [b,h,s,d]
__global__ void rope_pack_q(const float* __restrict__ Q,
                            const float* __restrict__ cosT,
                            const float* __restrict__ sinT,
                            unsigned short* __restrict__ Qbf) {
  int i = blockIdx.x * 256 + threadIdx.x;
  if (i >= BATCH * SEQ * NH * HD) return;
  int d = i & 63, h = (i >> 6) & 31, s = (i >> 11) & 2047, b = i >> 22;
  float x = Q[i];
  float other = (d < 32) ? -Q[i + 32] : Q[i - 32];
  float r = x * cosT[s * HD + d] + other * sinT[s * HD + d];
  Qbf[(((size_t)(b * NH + h) * SEQ + s) << 6) + d] = f2bf(r);
}

// RoPE on K (fp32, [b,s,kvh*64+d]) -> bf16 [b,kvh,s,d]
__global__ void rope_pack_k(const float* __restrict__ K,
                            const float* __restrict__ cosT,
                            const float* __restrict__ sinT,
                            unsigned short* __restrict__ Kbf) {
  int i = blockIdx.x * 256 + threadIdx.x;
  if (i >= BATCH * SEQ * NKV * HD) return;
  int d = i & 63, kvh = (i >> 6) & 7, s = (i >> 9) & 2047, b = i >> 20;
  float x = K[i];
  float other = (d < 32) ? -K[i + 32] : K[i - 32];
  float r = x * cosT[s * HD + d] + other * sinT[s * HD + d];
  Kbf[(((size_t)(b * NKV + kvh) * SEQ + s) << 6) + d] = f2bf(r);
}

// V (fp32, [b,s,kvh*64+d]) -> bf16 transposed [b,kvh,d,s]
__global__ void pack_vt(const float* __restrict__ V,
                        unsigned short* __restrict__ Vt) {
  int i = blockIdx.x * 256 + threadIdx.x;
  if (i >= BATCH * SEQ * NKV * HD) return;
  int d = i & 63, kvh = (i >> 6) & 7, s = (i >> 9) & 2047, b = i >> 20;
  Vt[((size_t)(b * NKV + kvh) * HD + d) * SEQ + s] = f2bf(V[i]);
}

// Pass A: dense-causal softmax row stats (m, l) per query. One wave = 16 queries.
__global__ void attn_stats(const unsigned short* __restrict__ Qbf,
                           const unsigned short* __restrict__ Kbf,
                           float* __restrict__ Ms, float* __restrict__ Ls) {
  int b = blockIdx.z, h = blockIdx.y, q0 = blockIdx.x * 16;
  const unsigned short* Qh = Qbf + (size_t)(b * NH + h) * SEQ * HD;
  const unsigned short* Kh = Kbf + (size_t)(b * NKV + (h >> 2)) * SEQ * HD;
  int lane = threadIdx.x & 31, half = lane >> 4, nl = lane & 15;
  float m[8], l[8];
#pragma unroll
  for (int r = 0; r < 8; ++r) { m[r] = NEGF; l[r] = 0.f; }
  for (int kb = 0; kb < q0 + 16; kb += 32) {
    v8f s0{}, s1{};
#pragma unroll
    for (int kk = 0; kk < HD; kk += 32) {
      v16bf a  = frag_a(Qh, HD, q0, kk);
      v16bf b0 = frag_b(Kh, HD, kb, kk);
      v16bf b1 = frag_b(Kh, HD, kb + 16, kk);
      s0 = wmma_bf16(a, b0, s0);
      s1 = wmma_bf16(a, b1, s1);
    }
#pragma unroll
    for (int r = 0; r < 8; ++r) {
      int qg = q0 + r + half * 8;
      float sc0 = (kb + nl      <= qg) ? s0[r] * SCALE : NEGF;
      float sc1 = (kb + 16 + nl <= qg) ? s1[r] * SCALE : NEGF;
      float v = fmaxf(sc0, sc1);
      v = fmaxf(v, __shfl_xor(v, 1, 16));
      v = fmaxf(v, __shfl_xor(v, 2, 16));
      v = fmaxf(v, __shfl_xor(v, 4, 16));
      v = fmaxf(v, __shfl_xor(v, 8, 16));
      float mn = fmaxf(m[r], v);
      float corr = __expf(m[r] - mn);
      float e = __expf(sc0 - mn) + __expf(sc1 - mn);
      e += __shfl_xor(e, 1, 16);
      e += __shfl_xor(e, 2, 16);
      e += __shfl_xor(e, 4, 16);
      e += __shfl_xor(e, 8, 16);
      l[r] = l[r] * corr + e;
      m[r] = mn;
    }
  }
  if (nl == 0) {
    size_t base = (size_t)(b * NH + h) * SEQ;
#pragma unroll
    for (int r = 0; r < 8; ++r) {
      Ms[base + q0 + r + half * 8] = m[r];
      Ls[base + q0 + r + half * 8] = l[r];
    }
  }
}

// Pass B: importance[k] = sum_q probs[q,k] — key-major, deterministic (no atomics).
__global__ void attn_importance(const unsigned short* __restrict__ Qbf,
                                const unsigned short* __restrict__ Kbf,
                                const float* __restrict__ Ms,
                                const float* __restrict__ Ls,
                                float* __restrict__ Imp) {
  int b = blockIdx.z, h = blockIdx.y, k0 = blockIdx.x * 16;
  const unsigned short* Qh = Qbf + (size_t)(b * NH + h) * SEQ * HD;
  const unsigned short* Kh = Kbf + (size_t)(b * NKV + (h >> 2)) * SEQ * HD;
  size_t sb = (size_t)(b * NH + h) * SEQ;
  int lane = threadIdx.x & 31, half = lane >> 4, nl = lane & 15;
  float acc[8];
#pragma unroll
  for (int r = 0; r < 8; ++r) acc[r] = 0.f;
  int qb0 = (k0 >> 5) << 5;
  for (int qb = qb0; qb < SEQ; qb += 32) {
    v8f s0{}, s1{};               // s[key, query] tiles (transposed scores)
#pragma unroll
    for (int kk = 0; kk < HD; kk += 32) {
      v16bf a  = frag_a(Kh, HD, k0, kk);
      v16bf b0 = frag_b(Qh, HD, qb, kk);
      v16bf b1 = frag_b(Qh, HD, qb + 16, kk);
      s0 = wmma_bf16(a, b0, s0);
      s1 = wmma_bf16(a, b1, s1);
    }
    float mq0 = Ms[sb + qb + nl],      li0 = 1.f / Ls[sb + qb + nl];
    float mq1 = Ms[sb + qb + 16 + nl], li1 = 1.f / Ls[sb + qb + 16 + nl];
#pragma unroll
    for (int r = 0; r < 8; ++r) {
      int kg = k0 + r + half * 8;
      float p0 = (qb + nl      >= kg) ? __expf(s0[r] * SCALE - mq0) * li0 : 0.f;
      float p1 = (qb + 16 + nl >= kg) ? __expf(s1[r] * SCALE - mq1) * li1 : 0.f;
      acc[r] += p0 + p1;
    }
  }
#pragma unroll
  for (int r = 0; r < 8; ++r) {
    float v = acc[r];
    v += __shfl_xor(v, 1, 16);
    v += __shfl_xor(v, 2, 16);
    v += __shfl_xor(v, 4, 16);
    v += __shfl_xor(v, 8, 16);
    if (nl == 0) Imp[sb + k0 + r + half * 8] = v;
  }
}

// Top-k(819) per (b,h): LDS bitonic sort of (value,index), emit byte mask.
__global__ void topk_mask(const float* __restrict__ Imp,
                          unsigned char* __restrict__ Msk) {
  __shared__ float val[SEQ];
  __shared__ short idx[SEQ];
  int bh = blockIdx.x, tid = threadIdx.x;
  for (int i = tid; i < SEQ; i += 256) {
    val[i] = (i < 4) ? 3.0e38f : Imp[(size_t)bh * SEQ + i];
    idx[i] = (short)i;
  }
  __syncthreads();
  for (int k = 2; k <= SEQ; k <<= 1) {
    for (int j = k >> 1; j > 0; j >>= 1) {
      for (int i = tid; i < SEQ; i += 256) {
        int ixj = i ^ j;
        if (ixj > i) {
          bool up = ((i & k) == 0);            // descending overall
          float a = val[i], c = val[ixj];
          if (up ? (a < c) : (a > c)) {
            val[i] = c; val[ixj] = a;
            short t = idx[i]; idx[i] = idx[ixj]; idx[ixj] = t;
          }
        }
      }
      __syncthreads();
    }
  }
  for (int i = tid; i < SEQ; i += 256) Msk[(size_t)bh * SEQ + i] = 0;
  __syncthreads();
  for (int i = tid; i < KEEP; i += 256) Msk[(size_t)bh * SEQ + idx[i]] = 1;
}

// Pass C: sparse-masked flash attention; P goes C-layout -> A-layout via LDS.
__global__ void attn_sparse(const unsigned short* __restrict__ Qbf,
                            const unsigned short* __restrict__ Kbf,
                            const unsigned short* __restrict__ Vt,
                            const unsigned char* __restrict__ Msk,
                            unsigned short* __restrict__ Obf) {
  __shared__ unsigned short Pt[16 * 32];
  int b = blockIdx.z, h = blockIdx.y, q0 = blockIdx.x * 16;
  const unsigned short* Qh = Qbf + (size_t)(b * NH + h) * SEQ * HD;
  const unsigned short* Kh = Kbf + (size_t)(b * NKV + (h >> 2)) * SEQ * HD;
  const unsigned short* Vh = Vt  + (size_t)(b * NKV + (h >> 2)) * HD * SEQ;
  const unsigned char*  Mh = Msk + (size_t)(b * NH + h) * SEQ;
  int lane = threadIdx.x & 31, half = lane >> 4, nl = lane & 15;
  float m[8], l[8];
  v8f o0{}, o1{}, o2{}, o3{};
#pragma unroll
  for (int r = 0; r < 8; ++r) { m[r] = NEGF; l[r] = 0.f; }

  for (int kb = 0; kb < q0 + 16; kb += 32) {
    v8f s0{}, s1{};
#pragma unroll
    for (int kk = 0; kk < HD; kk += 32) {
      v16bf a  = frag_a(Qh, HD, q0, kk);
      v16bf b0 = frag_b(Kh, HD, kb, kk);
      v16bf b1 = frag_b(Kh, HD, kb + 16, kk);
      s0 = wmma_bf16(a, b0, s0);
      s1 = wmma_bf16(a, b1, s1);
    }
    int tk0 = Mh[kb + nl];
    int tk1 = Mh[kb + 16 + nl];
#pragma unroll
    for (int r = 0; r < 8; ++r) {
      int qg = q0 + r + half * 8;
      float sc0 = ((kb + nl      <= qg) && tk0) ? s0[r] * SCALE : NEGF;
      float sc1 = ((kb + 16 + nl <= qg) && tk1) ? s1[r] * SCALE : NEGF;
      float v = fmaxf(sc0, sc1);
      v = fmaxf(v, __shfl_xor(v, 1, 16));
      v = fmaxf(v, __shfl_xor(v, 2, 16));
      v = fmaxf(v, __shfl_xor(v, 4, 16));
      v = fmaxf(v, __shfl_xor(v, 8, 16));
      float mn = fmaxf(m[r], v);
      float corr = __expf(m[r] - mn);
      float e0 = __expf(sc0 - mn), e1 = __expf(sc1 - mn);
      float e = e0 + e1;
      e += __shfl_xor(e, 1, 16);
      e += __shfl_xor(e, 2, 16);
      e += __shfl_xor(e, 4, 16);
      e += __shfl_xor(e, 8, 16);
      l[r] = l[r] * corr + e;
      m[r] = mn;
      o0[r] *= corr; o1[r] *= corr; o2[r] *= corr; o3[r] *= corr;
      Pt[(r + half * 8) * 32 + nl]      = f2bf(e0);
      Pt[(r + half * 8) * 32 + 16 + nl] = f2bf(e1);
    }
    __syncthreads();
    FragU pa;
    pa.q[0] = *(const uint4*)&Pt[nl * 32 + half * 8];
    pa.q[1] = *(const uint4*)&Pt[nl * 32 + half * 8 + 16];
#pragma unroll
    for (int nd = 0; nd < 4; ++nd) {
      FragU fb;
      const unsigned short* pv = Vh + (size_t)(nd * 16 + nl) * SEQ + kb + half * 8;
      fb.q[0] = *(const uint4*)(pv);
      fb.q[1] = *(const uint4*)(pv + 16);
      if (nd == 0) o0 = wmma_bf16(pa.v, fb.v, o0);
      if (nd == 1) o1 = wmma_bf16(pa.v, fb.v, o1);
      if (nd == 2) o2 = wmma_bf16(pa.v, fb.v, o2);
      if (nd == 3) o3 = wmma_bf16(pa.v, fb.v, o3);
    }
    __syncthreads();
  }
  size_t ob = (size_t)b * SEQ * (NH * HD);
#pragma unroll
  for (int r = 0; r < 8; ++r) {
    int qg = q0 + r + half * 8;
    float inv = 1.f / l[r];
    size_t row = ob + (size_t)qg * (NH * HD) + h * HD;
    Obf[row +  0 + nl] = f2bf(o0[r] * inv);
    Obf[row + 16 + nl] = f2bf(o1[r] * inv);
    Obf[row + 32 + nl] = f2bf(o2[r] * inv);
    Obf[row + 48 + nl] = f2bf(o3[r] * inv);
  }
}

// ---------------------------------------------------------------- launcher
extern "C" void kernel_launch(void* const* d_in, const int* in_sizes, int n_in,
                              void* d_out, int out_size, void* d_ws, size_t ws_size,
                              hipStream_t stream) {
  (void)in_sizes; (void)n_in; (void)out_size; (void)ws_size;
  const float* hs   = (const float*)d_in[0];
  const float* Wq   = (const float*)d_in[1];
  const float* Wk   = (const float*)d_in[2];
  const float* Wv   = (const float*)d_in[3];
  const float* Wo   = (const float*)d_in[4];
  const float* cosT = (const float*)d_in[5];
  const float* sinT = (const float*)d_in[6];
  float* out = (float*)d_out;

  char* w = (char*)d_ws;
  size_t off = 0;
  auto carve = [&](size_t bytes) -> char* {
    char* p = w + off; off += (bytes + 255) & ~(size_t)255; return p;
  };
  const size_t NT = (size_t)BATCH * SEQ;           // 4096 token rows
  unsigned short* hsb = (unsigned short*)carve(NT * HIDDEN * 2);
  unsigned short* wqb = (unsigned short*)carve((size_t)HIDDEN * HIDDEN * 2);
  unsigned short* wkb = (unsigned short*)carve((size_t)NKV * HD * HIDDEN * 2);
  unsigned short* wvb = (unsigned short*)carve((size_t)NKV * HD * HIDDEN * 2);
  unsigned short* wob = (unsigned short*)carve((size_t)HIDDEN * HIDDEN * 2);
  float* Qf = (float*)carve(NT * HIDDEN * 4);
  float* Kf = (float*)carve(NT * NKV * HD * 4);
  float* Vf = (float*)carve(NT * NKV * HD * 4);
  unsigned short* Qbf = (unsigned short*)carve((size_t)BATCH * NH * SEQ * HD * 2);
  unsigned short* Kbf = (unsigned short*)carve((size_t)BATCH * NKV * SEQ * HD * 2);
  unsigned short* Vtb = (unsigned short*)carve((size_t)BATCH * NKV * HD * SEQ * 2);
  float* Ms  = (float*)carve((size_t)BATCH * NH * SEQ * 4);
  float* Ls  = (float*)carve((size_t)BATCH * NH * SEQ * 4);
  float* Imp = (float*)carve((size_t)BATCH * NH * SEQ * 4);
  unsigned char* Msk = (unsigned char*)carve((size_t)BATCH * NH * SEQ);
  // Obf reuses Qf's region (Qf is dead after rope_pack_q; stream-ordered).
  unsigned short* Obf = (unsigned short*)Qf;

  int n;
  n = (int)(NT * HIDDEN);
  f32_to_bf16_k<<<(n + 255) / 256, 256, 0, stream>>>(hs, hsb, n);
  n = HIDDEN * HIDDEN;
  f32_to_bf16_k<<<(n + 255) / 256, 256, 0, stream>>>(Wq, wqb, n);
  n = NKV * HD * HIDDEN;
  f32_to_bf16_k<<<(n + 255) / 256, 256, 0, stream>>>(Wk, wkb, n);
  f32_to_bf16_k<<<(n + 255) / 256, 256, 0, stream>>>(Wv, wvb, n);
  n = HIDDEN * HIDDEN;
  f32_to_bf16_k<<<(n + 255) / 256, 256, 0, stream>>>(Wo, wob, n);

  dim3 blk128(128);
  gemm_xyt_lds<<<dim3(HIDDEN / 64, (int)NT / 128), blk128, 0, stream>>>(
      hsb, wqb, Qf, (int)NT, HIDDEN, HIDDEN);
  gemm_xyt_lds<<<dim3(NKV * HD / 64, (int)NT / 128), blk128, 0, stream>>>(
      hsb, wkb, Kf, (int)NT, NKV * HD, HIDDEN);
  gemm_xyt_lds<<<dim3(NKV * HD / 64, (int)NT / 128), blk128, 0, stream>>>(
      hsb, wvb, Vf, (int)NT, NKV * HD, HIDDEN);

  n = (int)(NT * HIDDEN);
  rope_pack_q<<<(n + 255) / 256, 256, 0, stream>>>(Qf, cosT, sinT, Qbf);
  n = (int)(NT * NKV * HD);
  rope_pack_k<<<(n + 255) / 256, 256, 0, stream>>>(Kf, cosT, sinT, Kbf);
  pack_vt<<<(n + 255) / 256, 256, 0, stream>>>(Vf, Vtb);

  dim3 wv32(32);
  dim3 ag(SEQ / 16, NH, BATCH);
  attn_stats<<<ag, wv32, 0, stream>>>(Qbf, Kbf, Ms, Ls);
  attn_importance<<<ag, wv32, 0, stream>>>(Qbf, Kbf, Ms, Ls, Imp);
  topk_mask<<<BATCH * NH, 256, 0, stream>>>(Imp, Msk);
  attn_sparse<<<ag, wv32, 0, stream>>>(Qbf, Kbf, Vtb, Msk, Obf);

  gemm_xyt_lds<<<dim3(HIDDEN / 64, (int)NT / 128), blk128, 0, stream>>>(
      Obf, wob, out, (int)NT, HIDDEN, NH * HD);
}
